// SwitchSAE_44229573214857
// MI455X (gfx1250) — matrix operand order repaired and checked by reading, not compiled, not run on hardware
//
#include <hip/hip_runtime.h>
#include <hip/hip_bf16.h>
#include <cstdint>

#define N_TOKENS   16384
#define N_EXPERTS  8
#define D_IN       768
#define EXPERT_DIM 2048

typedef __attribute__((ext_vector_type(16))) __bf16 v16bf;
typedef __attribute__((ext_vector_type(8)))  __bf16 v8bf;
typedef __attribute__((ext_vector_type(8)))  float  v8f;

#if __has_builtin(__builtin_amdgcn_global_load_async_to_lds_b128) && \
    __has_builtin(__builtin_amdgcn_s_wait_asynccnt)
#define USE_ASYNC_LDS 1
typedef int v4i_ls __attribute__((vector_size(4 * sizeof(int))));
#else
#define USE_ASYNC_LDS 0
#endif

__device__ __forceinline__ unsigned short f32_bf16(float f) {
  __bf16 h = (__bf16)f;                      // native RNE convert on gfx1250
  return __builtin_bit_cast(unsigned short, h);
}

__device__ __forceinline__ v16bf ld16(const unsigned short* p0, const unsigned short* p1) {
  v8bf a = *reinterpret_cast<const v8bf*>(p0);
  v8bf b = *reinterpret_cast<const v8bf*>(p1);
  return __builtin_shufflevector(a, b, 0,1,2,3,4,5,6,7,8,9,10,11,12,13,14,15);
}

// ---------------------------------------------------------------------------
// Router: one wave32 per token. logits = (x - router_b) @ router; softmax;
// top-1 prob + expert; atomic scatter into per-expert token lists.
// ---------------------------------------------------------------------------
__global__ __launch_bounds__(256) void sae_router(
    const float* __restrict__ act, const float* __restrict__ router_b,
    const float* __restrict__ router, float* __restrict__ prob,
    int* __restrict__ cnt, int* __restrict__ list)
{
  const int lane = threadIdx.x & 31;
  const int wave = threadIdx.x >> 5;
  const int t = blockIdx.x * 8 + wave;

  float acc[N_EXPERTS];
  #pragma unroll
  for (int e = 0; e < N_EXPERTS; ++e) acc[e] = 0.f;

  const float* x = act + (size_t)t * D_IN;
  for (int d = lane; d < D_IN; d += 32) {
    float xv = x[d] - router_b[d];
    const float* rrow = router + (size_t)d * N_EXPERTS;
    #pragma unroll
    for (int e = 0; e < N_EXPERTS; ++e) acc[e] += xv * rrow[e];
  }
  #pragma unroll
  for (int e = 0; e < N_EXPERTS; ++e) {
    #pragma unroll
    for (int off = 16; off > 0; off >>= 1)
      acc[e] += __shfl_xor(acc[e], off, 32);
  }
  if (lane == 0) {
    float m = acc[0]; int bi = 0;
    #pragma unroll
    for (int e = 1; e < N_EXPERTS; ++e)
      if (acc[e] > m) { m = acc[e]; bi = e; }   // first-max, matches argmax
    float s = 0.f;
    #pragma unroll
    for (int e = 0; e < N_EXPERTS; ++e) s += __expf(acc[e] - m);
    prob[t] = 1.0f / s;                         // = max softmax prob
    int pos = atomicAdd(&cnt[bi], 1);
    list[bi * N_TOKENS + pos] = t;
  }
}

// ---------------------------------------------------------------------------
// Encode: latent[t] = relu( act[t] @ enc[e] )  for gathered tokens of expert e
// 128x128 tile, K-step 32, bf16 WMMA with f32 accumulate.
// ---------------------------------------------------------------------------
__global__ __launch_bounds__(256) void sae_encode(
    const float* __restrict__ act, const float* __restrict__ enc,
    const int* __restrict__ cnt, const int* __restrict__ list,
    unsigned short* __restrict__ latent)
{
  const int e  = blockIdx.z;
  const int mt = blockIdx.x;
  const int nt = blockIdx.y;
  const int ne = cnt[e];
  if (mt * 128 >= ne) return;

  __shared__ __align__(16) unsigned short aT[128 * 32];   // [row][k]  bf16
  __shared__ __align__(16) unsigned short bT[32 * 128];   // [k][n]    bf16

  const int tid  = threadIdx.x;
  const int lane = tid & 31;
  const int wave = tid >> 5;
  const int half = lane >> 4;
  const int l16  = lane & 15;

  const int* mylist = list + e * N_TOKENS + mt * 128;
  const float* encE = enc + (size_t)e * D_IN * EXPERT_DIM + (size_t)nt * 128;

  // Hoist K-invariant row -> token mapping for the A fill (rows tid>>3 + p*32).
  int arow_t[4]; bool arow_v[4];
  #pragma unroll
  for (int p = 0; p < 4; ++p) {
    int r = p * 32 + (tid >> 3);
    arow_v[p] = (mt * 128 + r < ne);
    arow_t[p] = arow_v[p] ? mylist[r] : 0;
  }

  v8f acc[8];
  #pragma unroll
  for (int j = 0; j < 8; ++j) acc[j] = (v8f){0.f,0.f,0.f,0.f,0.f,0.f,0.f,0.f};

  for (int k0 = 0; k0 < D_IN; k0 += 32) {
    __syncthreads();
    // A tile: gathered activation rows, fp32 -> bf16
    #pragma unroll
    for (int p = 0; p < 4; ++p) {
      int r = p * 32 + (tid >> 3);
      int c = (tid & 7) * 4;
      float4 v = make_float4(0.f, 0.f, 0.f, 0.f);
      if (arow_v[p])
        v = *reinterpret_cast<const float4*>(act + (size_t)arow_t[p] * D_IN + k0 + c);
      unsigned short* dst = &aT[r * 32 + c];
      dst[0] = f32_bf16(v.x); dst[1] = f32_bf16(v.y);
      dst[2] = f32_bf16(v.z); dst[3] = f32_bf16(v.w);
    }
    // B tile: enc[e][k0..k0+31][nt*128 .. +127], fp32 -> bf16
    #pragma unroll
    for (int p = 0; p < 4; ++p) {
      int kk = p * 8 + (tid >> 5);
      int c  = (tid & 31) * 4;
      const float* src = encE + (size_t)(k0 + kk) * EXPERT_DIM + c;
      float4 v = *reinterpret_cast<const float4*>(src);
      if (k0 + 32 < D_IN) __builtin_prefetch(src + 32 * EXPERT_DIM, 0, 1);
      unsigned short* dst = &bT[kk * 128 + c];
      dst[0] = f32_bf16(v.x); dst[1] = f32_bf16(v.y);
      dst[2] = f32_bf16(v.z); dst[3] = f32_bf16(v.w);
    }
    __syncthreads();
    // A operand: lane holds row M=l16, K = half*8..+7 and half*8+16..+23
    const unsigned short* ap = &aT[(wave * 16 + l16) * 32 + half * 8];
    v16bf a = ld16(ap, ap + 16);
    // Preload all 8 B operands, then issue 8 back-to-back WMMAs (single wait).
    v16bf b[8];
    #pragma unroll
    for (int j = 0; j < 8; ++j) {
      const unsigned short* bp = &bT[half * 8 * 128 + j * 16 + l16];
      #pragma unroll
      for (int s = 0; s < 16; ++s)
        b[j][s] = __builtin_bit_cast(__bf16, bp[((s & 7) + (s >> 3) * 16) * 128]);
    }
    #pragma unroll
    for (int j = 0; j < 8; ++j)
      acc[j] = __builtin_amdgcn_wmma_f32_16x16x32_bf16(
          false, a, false, b[j], (short)0, acc[j], false, false);
  }

  // Epilogue: ReLU, bf16, scatter by token id. D: VGPR r -> M = r + 8*half.
  #pragma unroll
  for (int r = 0; r < 8; ++r) {
    int lr = wave * 16 + r + 8 * half;
    if (mt * 128 + lr < ne) {
      int t = mylist[lr];
      unsigned short* orow = latent + (size_t)t * EXPERT_DIM + nt * 128 + l16;
      #pragma unroll
      for (int j = 0; j < 8; ++j) {
        float v = acc[j][r];
        v = v > 0.f ? v : 0.f;
        orow[j * 16] = f32_bf16(v);
      }
    }
  }
}

// ---------------------------------------------------------------------------
// Decode: out[t] = prob[t] * ( latent[t] @ dec[e] ) + pre_b
// A tile (bf16 latent) is copied via async global->LDS when available.
// ---------------------------------------------------------------------------
__global__ __launch_bounds__(256) void sae_decode(
    const unsigned short* __restrict__ latent, const float* __restrict__ dec,
    const float* __restrict__ pre_b, const float* __restrict__ prob,
    const int* __restrict__ cnt, const int* __restrict__ list,
    float* __restrict__ out)
{
  const int e  = blockIdx.z;
  const int mt = blockIdx.x;
  const int nt = blockIdx.y;
  const int ne = cnt[e];
  if (mt * 128 >= ne) return;

  __shared__ __align__(16) unsigned short aT[128 * 32];
  __shared__ __align__(16) unsigned short bT[32 * 128];

  const int tid  = threadIdx.x;
  const int lane = tid & 31;
  const int wave = tid >> 5;
  const int half = lane >> 4;
  const int l16  = lane & 15;

  const int* mylist = list + e * N_TOKENS + mt * 128;
  const float* decE = dec + (size_t)e * EXPERT_DIM * D_IN + (size_t)nt * 128;

  // A tile copy plan: 128 rows x 64B = 512 x 16B chunks; 2 chunks per thread.
  // chunk = p*256 + tid -> row = chunk>>2, 16B sub-chunk = (chunk&3)*8 elems.
  int  crow_t[2]; bool crow_v[2]; int crow_r[2], csub[2];
  #pragma unroll
  for (int p = 0; p < 2; ++p) {
    int chunk = p * 256 + tid;
    crow_r[p] = chunk >> 2;
    csub[p]   = (chunk & 3) * 8;
    crow_v[p] = (mt * 128 + crow_r[p] < ne);
    crow_t[p] = crow_v[p] ? mylist[crow_r[p]] : 0;
  }

  v8f acc[8];
  #pragma unroll
  for (int j = 0; j < 8; ++j) acc[j] = (v8f){0.f,0.f,0.f,0.f,0.f,0.f,0.f,0.f};

  for (int k0 = 0; k0 < EXPERT_DIM; k0 += 32) {
    __syncthreads();
    // ---- A tile: latent rows (bf16) gathered by token id
    #pragma unroll
    for (int p = 0; p < 2; ++p) {
      unsigned short* ldst = &aT[crow_r[p] * 32 + csub[p]];
      if (crow_v[p]) {
        const unsigned short* gsrc = latent + (size_t)crow_t[p] * EXPERT_DIM + k0 + csub[p];
#if USE_ASYNC_LDS
        __builtin_amdgcn_global_load_async_to_lds_b128(
            (v4i_ls*)gsrc, (v4i_ls*)ldst, 0, 0);
#else
        *reinterpret_cast<uint4*>(ldst) = *reinterpret_cast<const uint4*>(gsrc);
#endif
      } else {
        *reinterpret_cast<uint4*>(ldst) = make_uint4(0u, 0u, 0u, 0u);
      }
    }
    // ---- B tile: dec[e][k0..k0+31][nt*128 .. +127], fp32 -> bf16
    #pragma unroll
    for (int p = 0; p < 4; ++p) {
      int kk = p * 8 + (tid >> 5);
      int c  = (tid & 31) * 4;
      const float* src = decE + (size_t)(k0 + kk) * D_IN + c;
      float4 v = *reinterpret_cast<const float4*>(src);
      if (k0 + 32 < EXPERT_DIM) __builtin_prefetch(src + 32 * D_IN, 0, 1);
      unsigned short* dst = &bT[kk * 128 + c];
      dst[0] = f32_bf16(v.x); dst[1] = f32_bf16(v.y);
      dst[2] = f32_bf16(v.z); dst[3] = f32_bf16(v.w);
    }
#if USE_ASYNC_LDS
    __builtin_amdgcn_s_wait_asynccnt(0);   // my async LDS writes done before barrier
#endif
    __syncthreads();
    const unsigned short* ap = &aT[(wave * 16 + l16) * 32 + half * 8];
    v16bf a = ld16(ap, ap + 16);
    v16bf b[8];
    #pragma unroll
    for (int j = 0; j < 8; ++j) {
      const unsigned short* bp = &bT[half * 8 * 128 + j * 16 + l16];
      #pragma unroll
      for (int s = 0; s < 16; ++s)
        b[j][s] = __builtin_bit_cast(__bf16, bp[((s & 7) + (s >> 3) * 16) * 128]);
    }
    #pragma unroll
    for (int j = 0; j < 8; ++j)
      acc[j] = __builtin_amdgcn_wmma_f32_16x16x32_bf16(
          false, a, false, b[j], (short)0, acc[j], false, false);
  }

  float pb[8];
  #pragma unroll
  for (int j = 0; j < 8; ++j) pb[j] = pre_b[nt * 128 + j * 16 + l16];

  #pragma unroll
  for (int r = 0; r < 8; ++r) {
    int lr = wave * 16 + r + 8 * half;
    if (mt * 128 + lr < ne) {
      int t = mylist[lr];
      float ps = prob[t];
      float* orow = out + (size_t)t * D_IN + nt * 128 + l16;
      #pragma unroll
      for (int j = 0; j < 8; ++j)
        orow[j * 16] = ps * acc[j][r] + pb[j];
    }
  }
}

// ---------------------------------------------------------------------------
extern "C" void kernel_launch(void* const* d_in, const int* in_sizes, int n_in,
                              void* d_out, int out_size, void* d_ws, size_t ws_size,
                              hipStream_t stream) {
  const float* act      = (const float*)d_in[0];
  const float* pre_b    = (const float*)d_in[1];
  const float* enc      = (const float*)d_in[2];
  const float* dec      = (const float*)d_in[3];
  const float* router_b = (const float*)d_in[4];
  const float* router   = (const float*)d_in[5];
  float* out = (float*)d_out;

  // Workspace layout (all offsets 256B aligned):
  //   [0,32)                 expert counters (8 x i32)
  //   [256, 256+64K)         prob[N_TOKENS]  f32
  //   [+, +512K)             token lists [E][N_TOKENS] i32
  //   [+, +64MB)             latent [N_TOKENS][EXPERT_DIM] bf16
  char* ws = (char*)d_ws;
  int*   cnt  = (int*)(ws);
  float* prob = (float*)(ws + 256);
  int*   list = (int*)(ws + 256 + N_TOKENS * sizeof(float));
  unsigned short* latent =
      (unsigned short*)(ws + 256 + N_TOKENS * sizeof(float)
                           + (size_t)N_EXPERTS * N_TOKENS * sizeof(int));

  (void)hipMemsetAsync(cnt, 0, N_EXPERTS * sizeof(int), stream);

  sae_router<<<N_TOKENS / 8, 256, 0, stream>>>(act, router_b, router, prob, cnt, list);

  sae_encode<<<dim3(N_TOKENS / 128, EXPERT_DIM / 128, N_EXPERTS), 256, 0, stream>>>(
      act, enc, cnt, list, latent);

  sae_decode<<<dim3(N_TOKENS / 128, D_IN / 128, N_EXPERTS), 256, 0, stream>>>(
      latent, dec, pre_b, prob, cnt, list, out);
}
